// VqVae_31645319037291
// MI455X (gfx1250) — compile-verified, matrix operand order checked by reading, not run on hardware
//
#include <hip/hip_runtime.h>
#include <hip/hip_bf16.h>

typedef __bf16 bf16;
typedef __attribute__((ext_vector_type(16))) __bf16 v16bf;
typedef __attribute__((ext_vector_type(8)))  __bf16 v8bf;
typedef __attribute__((ext_vector_type(8)))  float  v8f;

#define B_TOTAL   131072
#define TILE_ROWS 64
#define NBLK      (B_TOTAL / TILE_ROWS)   // 2048
#define NW        4                       // waves per block (wave32)

// ---------------------------------------------------------------------------
// WMMA helpers
// ---------------------------------------------------------------------------
__device__ __forceinline__ v8f wmma_bf16(v16bf a, v16bf b, v8f c) {
  return __builtin_amdgcn_wmma_f32_16x16x32_bf16(
      /*neg_a=*/false, a, /*neg_b=*/false, b,
      /*c_mod=*/(short)0, c, /*reuse_a=*/false, /*reuse_b=*/false);
}

// A fragment 16x32 (bf16): lanes 0-15 -> row=lane, K k0..k0+15;
// lanes 16-31 -> row=lane-16, K k0+16..k0+31. 16 contiguous bf16 per lane.
__device__ __forceinline__ v16bf ldA(const bf16* base, int width, int k0, int lane) {
  const bf16* p = base + (lane & 15) * width + k0 + ((lane >> 4) << 4);
  return *(const v16bf*)p;
}
// B fragment 32x16 (bf16) from [N][Kpad] (column-major weight): lane picks col.
__device__ __forceinline__ v16bf ldB(const bf16* base, int kpad, int n0, int k0, int lane) {
  const bf16* p = base + (n0 + (lane & 15)) * kpad + k0 + ((lane >> 4) << 4);
  return *(const v16bf*)p;
}

// C layout: reg v, lane l: row = v + 8*(l>>4), col = n0 + (l&15)

// One GEMM stage: C[16][N] = act( A[16][K] @ W[N][K]^T + bias ), bf16 out to LDS
// K<=128: hoist all A fragments out of the N loop.
// K>128 (K=512, N<=128): chunk K by 128, keep all N-tile accumulators live.
template<int K, int N, bool RELU>
__device__ __forceinline__ void gemm_bias(const bf16* A, const bf16* Bw,
                                          const float* __restrict__ bias,
                                          bf16* C, int lane) {
  const int col16 = lane & 15;
  const int rb    = (lane >> 4) << 3;
  if constexpr (K <= 128) {
    v16bf afr[K / 32];
#pragma unroll
    for (int k = 0; k < K / 32; ++k) afr[k] = ldA(A, K, k * 32, lane);
    for (int n0 = 0; n0 < N; n0 += 16) {
      v8f acc = {};
#pragma unroll
      for (int k = 0; k < K / 32; ++k)
        acc = wmma_bf16(afr[k], ldB(Bw, K, n0, k * 32, lane), acc);
      const int col = n0 + col16;
      const float bv = bias[col];
#pragma unroll
      for (int v = 0; v < 8; ++v) {
        float x = acc[v] + bv;
        if (RELU) x = fmaxf(x, 0.f);
        C[(rb + v) * N + col] = (bf16)x;
      }
    }
  } else {
    static_assert(N <= 128, "persistent-acc path requires N<=128");
    v8f acc[N / 16];
#pragma unroll
    for (int t = 0; t < N / 16; ++t) { v8f z = {}; acc[t] = z; }
    for (int kc = 0; kc < K; kc += 128) {
      v16bf afr[4];
#pragma unroll
      for (int k = 0; k < 4; ++k) afr[k] = ldA(A, K, kc + k * 32, lane);
#pragma unroll
      for (int t = 0; t < N / 16; ++t)
#pragma unroll
        for (int k = 0; k < 4; ++k)
          acc[t] = wmma_bf16(afr[k], ldB(Bw, K, t * 16, kc + k * 32, lane), acc[t]);
    }
#pragma unroll
    for (int t = 0; t < N / 16; ++t) {
      const int col = t * 16 + col16;
      const float bv = bias[col];
#pragma unroll
      for (int v = 0; v < 8; ++v) {
        float x = acc[t][v] + bv;
        if (RELU) x = fmaxf(x, 0.f);
        C[(rb + v) * N + col] = (bf16)x;
      }
    }
  }
}

// ---------------------------------------------------------------------------
// Kernel 1: weight prep -> bf16 transposed/padded layouts in workspace
// ---------------------------------------------------------------------------
__global__ void vqvae_prep(const float* __restrict__ ew1, const float* __restrict__ ew2,
                           const float* __restrict__ ewf,
                           const float* __restrict__ dw1, const float* __restrict__ dw2,
                           const float* __restrict__ dwf,
                           const float* __restrict__ cb,
                           bf16* __restrict__ w1t, bf16* __restrict__ w2t,
                           bf16* __restrict__ wft,
                           bf16* __restrict__ d1t, bf16* __restrict__ d2t,
                           bf16* __restrict__ dft,
                           bf16* __restrict__ cbb, float* __restrict__ cnorm) {
  const int gsz = gridDim.x * blockDim.x;
  const int t0  = blockIdx.x * blockDim.x + threadIdx.x;
  // enc_w1 [90][128] -> w1t [128][96] (K padded to 96 with zeros)
  for (int i = t0; i < 128 * 96; i += gsz) {
    int n = i / 96, k = i - n * 96;
    w1t[i] = (bf16)((k < 90) ? ew1[k * 128 + n] : 0.f);
  }
  // enc_w2 [128][128] -> w2t [128][128]
  for (int i = t0; i < 128 * 128; i += gsz) {
    int n = i >> 7, k = i & 127;
    w2t[i] = (bf16)ew2[k * 128 + n];
  }
  // enc_wf [128][512] -> wft [512][128]
  for (int i = t0; i < 512 * 128; i += gsz) {
    int n = i >> 7, k = i & 127;
    wft[i] = (bf16)ewf[k * 512 + n];
  }
  // dec_w1 [512][128] -> d1t [128][512]
  for (int i = t0; i < 128 * 512; i += gsz) {
    int n = i >> 9, k = i & 511;
    d1t[i] = (bf16)dw1[k * 128 + n];
  }
  // dec_w2 [128][128] -> d2t [128][128]
  for (int i = t0; i < 128 * 128; i += gsz) {
    int n = i >> 7, k = i & 127;
    d2t[i] = (bf16)dw2[k * 128 + n];
  }
  // dec_wf [128][90] -> dft [96][128] (N padded to 96 with zeros)
  for (int i = t0; i < 96 * 128; i += gsz) {
    int n = i >> 7, k = i & 127;
    dft[i] = (bf16)((n < 90) ? dwf[k * 90 + n] : 0.f);
  }
  // codebooks [4][32][512] -> bf16 copy (already [code][D] = B-matrix layout)
  for (int i = t0; i < 4 * 32 * 512; i += gsz) cbb[i] = (bf16)cb[i];
  // per-code squared norms
  for (int i = t0; i < 4 * 32; i += gsz) {
    float s = 0.f;
    for (int d = 0; d < 512; ++d) { float v = cb[i * 512 + d]; s += v * v; }
    cnorm[i] = s;
  }
}

// ---------------------------------------------------------------------------
// Kernel 2: fused encoder -> residual VQ -> decoder -> loss partials
// ---------------------------------------------------------------------------
__global__ __launch_bounds__(NW * 32, 1)
void vqvae_fused(const float* __restrict__ state,
                 const float* __restrict__ eb1, const float* __restrict__ eb2,
                 const float* __restrict__ ebf,
                 const float* __restrict__ db1, const float* __restrict__ db2,
                 const float* __restrict__ dbf,
                 const bf16* __restrict__ w1t, const bf16* __restrict__ w2t,
                 const bf16* __restrict__ wft,
                 const bf16* __restrict__ d1t, const bf16* __restrict__ d2t,
                 const bf16* __restrict__ dft,
                 const bf16* __restrict__ cbbf, const float* __restrict__ cnorm,
                 float* __restrict__ partials, float* __restrict__ out) {
  __shared__ bf16  s_xb[NW][16 * 96];    // padded input tile (kept for losses)
  __shared__ bf16  s_ha[NW][16 * 128];   // hidden ping
  __shared__ bf16  s_hb[NW][16 * 128];   // hidden pong
  __shared__ bf16  s_rep[NW][16 * 512];  // encoder output -> residual (in place)
  __shared__ bf16  s_qs[NW][16 * 512];   // accumulated quantized sum
  __shared__ float s_sc[NW][16 * 32];    // VQ scores for argmin
  __shared__ int   s_ix[NW][16];         // per-row selected code
  __shared__ float s_rn[NW][16];         // per-row ||r||^2
  __shared__ float s_red[NW][3];         // block loss reduction

  const int tid  = threadIdx.x;
  const int w    = tid >> 5;
  const int lane = tid & 31;
  const long rowsBase = (long)blockIdx.x * TILE_ROWS + (long)w * 16;

  bf16*  xb  = s_xb[w];
  bf16*  ha  = s_ha[w];
  bf16*  hb  = s_hb[w];
  bf16*  rp  = s_rep[w];
  bf16*  qs  = s_qs[w];
  float* sc  = s_sc[w];
  int*   ix  = s_ix[w];
  float* rnb = s_rn[w];

  // warm L2 with the first-stage weights / codebooks (global_prefetch_b8)
  if (tid < 96)  __builtin_prefetch(w1t + tid * 128, 0, 1);
  if (tid < 128) {
    __builtin_prefetch(w2t + tid * 128, 0, 1);
    __builtin_prefetch(cbbf + tid * 512, 0, 1);
  }

  // ---- stage 0: load x (scale 1.0), pad 90->96, zero qsum -----------------
#pragma unroll
  for (int r = 0; r < 16; ++r)
    for (int c = lane; c < 96; c += 32) {
      float v = (c < 90) ? state[(rowsBase + r) * 90 + c] : 0.f;
      xb[r * 96 + c] = (bf16)v;
    }
  {
    uint4* q4 = (uint4*)qs;
    const uint4 z = make_uint4(0u, 0u, 0u, 0u);
    for (int e = lane; e < (16 * 512 * 2) / 16; e += 32) q4[e] = z;
  }

  // ---- encoder MLP ---------------------------------------------------------
  gemm_bias<96, 128, true >(xb, w1t, eb1, ha, lane);
  gemm_bias<128, 128, true >(ha, w2t, eb2, hb, lane);
  gemm_bias<128, 512, false>(hb, wft, ebf, rp, lane);

  // ---- residual VQ (4 groups, 32 codes, D=512) ----------------------------
  float vqS = 0.f;

  // initial ||r||^2 per row (vectorized; lanes 0-15 / 16-31 split columns)
  {
    const int row = lane & 15, half = lane >> 4;
    float rn = 0.f;
    const bf16* rr = rp + row * 512 + half * 256;
    for (int c = 0; c < 256; c += 8) {
      v8bf rv = *(const v8bf*)(rr + c);
#pragma unroll
      for (int i = 0; i < 8; ++i) { float f = (float)rv[i]; rn += f * f; }
    }
    rn += __shfl_xor(rn, 16);
    if (lane < 16) rnb[lane] = rn;
  }

  for (int g = 0; g < 4; ++g) {
    const bf16* cbg = cbbf + g * (32 * 512);
    const int row = lane & 15, half = lane >> 4;

    // dot(r, codes) via WMMA: M=16 rows, N=32 codes, K=512
    v8f a0 = {}, a1 = {};
#pragma unroll
    for (int k0 = 0; k0 < 512; k0 += 32) {
      v16bf a  = ldA(rp, 512, k0, lane);
      v16bf b0 = ldB(cbg, 512, 0, k0, lane);
      v16bf b1 = ldB(cbg, 512, 16, k0, lane);
      a0 = wmma_bf16(a, b0, a0);
      a1 = wmma_bf16(a, b1, a1);
    }
    // score = ||c||^2 - 2 r.c  (||r||^2 is row-constant -> irrelevant to argmin)
    const float cn0 = cnorm[g * 32 + row];
    const float cn1 = cnorm[g * 32 + 16 + row];
    const int rb = half << 3;
#pragma unroll
    for (int v = 0; v < 8; ++v) {
      sc[(rb + v) * 32 + row]      = cn0 - 2.f * a0[v];
      sc[(rb + v) * 32 + 16 + row] = cn1 - 2.f * a1[v];
    }

    // per-row argmin (lanes 0-15, one row each)
    if (lane < 16) {
      const float* sr = sc + lane * 32;
      float best = sr[0];
      int bi = 0;
      for (int c = 1; c < 32; ++c) {
        float s = sr[c];
        if (s < best) { best = s; bi = c; }
      }
      vqS += rnb[lane] + best;                         // min ||r - c||^2
      ix[lane] = bi;
      out[1 + (rowsBase + lane) * 4 + g] = (float)bi;  // vq_code
    }

    // r -= q ; qsum += q ; fuse next group's ||r||^2 (2 lanes per row)
    {
      const int ur = lane >> 1, c0 = (lane & 1) * 256;
      const int qi = ix[ur];
      const bf16* qr = cbg + qi * 512 + c0;
      bf16* rr = rp + ur * 512 + c0;
      bf16* qq = qs + ur * 512 + c0;
      float rn2 = 0.f;
      for (int c = 0; c < 256; c += 8) {
        v8bf qv = *(const v8bf*)(qr + c);
        v8bf rv = *(const v8bf*)(rr + c);
        v8bf sv = *(const v8bf*)(qq + c);
#pragma unroll
        for (int i = 0; i < 8; ++i) {
          float q  = (float)qv[i];
          float nr = (float)rv[i] - q;
          rv[i] = (bf16)nr;
          float nrr = (float)rv[i];          // bf16-rounded new residual
          rn2 += nrr * nrr;
          sv[i] = (bf16)((float)sv[i] + q);
        }
        *(v8bf*)(rr + c) = rv;
        *(v8bf*)(qq + c) = sv;
      }
      if (g < 3) {
        rn2 += __shfl_xor(rn2, 1);
        if (!(lane & 1)) rnb[ur] = rn2;
      }
    }
  }

  // ---- decoder MLP ---------------------------------------------------------
  gemm_bias<512, 128, true>(qs, d1t, db1, ha, lane);
  gemm_bias<128, 128, true>(ha, d2t, db2, hb, lane);

  // final layer fused with loss (N padded to 96; pad cols are exactly zero on
  // both x and dec_out, so they contribute nothing)
  float absS = 0.f, sqS = 0.f;
  {
    v16bf afr[4];
#pragma unroll
    for (int k = 0; k < 4; ++k) afr[k] = ldA(hb, 128, k * 32, lane);
    for (int n0 = 0; n0 < 96; n0 += 16) {
      v8f acc = {};
#pragma unroll
      for (int k = 0; k < 4; ++k)
        acc = wmma_bf16(afr[k], ldB(dft, 128, n0, k * 32, lane), acc);
      const int col = n0 + (lane & 15);
      const float bv = (col < 90) ? dbf[col] : 0.f;
      const int rb = (lane >> 4) << 3;
#pragma unroll
      for (int v = 0; v < 8; ++v) {
        float dec = acc[v] + bv;
        float xv  = (float)xb[(rb + v) * 96 + col];
        float d   = xv - dec;
        absS += fabsf(d);
        sqS  += d * d;
      }
    }
  }

  // ---- deterministic block reduction -> per-block partials ----------------
#pragma unroll
  for (int m = 16; m >= 1; m >>= 1) {
    absS += __shfl_xor(absS, m);
    sqS  += __shfl_xor(sqS, m);
    vqS  += __shfl_xor(vqS, m);
  }
  if (lane == 0) {
    s_red[w][0] = absS;
    s_red[w][1] = sqS;
    s_red[w][2] = vqS;
  }
  __syncthreads();
  if (tid == 0) {
    float a = 0.f, s = 0.f, v = 0.f;
#pragma unroll
    for (int i = 0; i < NW; ++i) {
      a += s_red[i][0];
      s += s_red[i][1];
      v += s_red[i][2];
    }
    partials[blockIdx.x * 4 + 0] = a;
    partials[blockIdx.x * 4 + 1] = s;
    partials[blockIdx.x * 4 + 2] = v;
  }
}

// ---------------------------------------------------------------------------
// Kernel 3: deterministic final reduction + scalar outputs
// ---------------------------------------------------------------------------
__global__ void vqvae_finalize(const float* __restrict__ partials,
                               float* __restrict__ out) {
  __shared__ float red[256][3];
  float a = 0.f, s = 0.f, v = 0.f;
  for (int i = threadIdx.x; i < NBLK; i += 256) {
    a += partials[i * 4 + 0];
    s += partials[i * 4 + 1];
    v += partials[i * 4 + 2];
  }
  red[threadIdx.x][0] = a;
  red[threadIdx.x][1] = s;
  red[threadIdx.x][2] = v;
  __syncthreads();
  if (threadIdx.x == 0) {
    float A = 0.f, S = 0.f, V = 0.f;
    for (int i = 0; i < 256; ++i) { A += red[i][0]; S += red[i][1]; V += red[i][2]; }
    const float denomX = (float)B_TOTAL * 90.f;
    const float denomD = (float)B_TOTAL * 512.f;
    const float enc_loss = A / denomX;
    const float rec_loss = S / denomX;
    const float vq_loss  = V / denomD;   // sum over 4 groups, same denominator
    const long codeCount = (long)B_TOTAL * 4;
    out[0] = enc_loss * 1.0f + vq_loss * 5.0f;
    out[1 + codeCount + 0] = vq_loss;
    out[1 + codeCount + 1] = rec_loss;
    out[1 + codeCount + 2] = enc_loss;
  }
}

// ---------------------------------------------------------------------------
extern "C" void kernel_launch(void* const* d_in, const int* in_sizes, int n_in,
                              void* d_out, int out_size, void* d_ws, size_t ws_size,
                              hipStream_t stream) {
  const float* state = (const float*)d_in[0];
  const float* ew1   = (const float*)d_in[1];
  const float* eb1   = (const float*)d_in[2];
  const float* ew2   = (const float*)d_in[3];
  const float* eb2   = (const float*)d_in[4];
  const float* ewf   = (const float*)d_in[5];
  const float* ebf   = (const float*)d_in[6];
  const float* dw1   = (const float*)d_in[7];
  const float* db1   = (const float*)d_in[8];
  const float* dw2   = (const float*)d_in[9];
  const float* db2   = (const float*)d_in[10];
  const float* dwf   = (const float*)d_in[11];
  const float* dbf   = (const float*)d_in[12];
  const float* cb    = (const float*)d_in[13];

  char* ws = (char*)d_ws;
  bf16*  w1t  = (bf16*) (ws + 0);        // 128*96*2   = 24576
  bf16*  w2t  = (bf16*) (ws + 24576);    // 128*128*2  = 32768
  bf16*  wft  = (bf16*) (ws + 57344);    // 512*128*2  = 131072
  bf16*  d1t  = (bf16*) (ws + 188416);   // 128*512*2  = 131072
  bf16*  d2t  = (bf16*) (ws + 319488);   // 128*128*2  = 32768
  bf16*  dft  = (bf16*) (ws + 352256);   // 96*128*2   = 24576
  bf16*  cbb  = (bf16*) (ws + 376832);   // 4*32*512*2 = 131072
  float* cn   = (float*)(ws + 507904);   // 128*4      = 512
  float* part = (float*)(ws + 508416);   // 2048*4*4   = 32768
  float* out  = (float*)d_out;

  vqvae_prep<<<256, 256, 0, stream>>>(ew1, ew2, ewf, dw1, dw2, dwf, cb,
                                      w1t, w2t, wft, d1t, d2t, dft, cbb, cn);
  vqvae_fused<<<NBLK, NW * 32, 0, stream>>>(state, eb1, eb2, ebf, db1, db2, dbf,
                                            w1t, w2t, wft, d1t, d2t, dft,
                                            cbb, cn, part, out);
  vqvae_finalize<<<1, 256, 0, stream>>>(part, out);
}
